// Pointnet2Backbone_69483980915296
// MI455X (gfx1250) — matrix operand order repaired and checked by reading, not compiled, not run on hardware
//
#include <hip/hip_runtime.h>
#include <hip/hip_bf16.h>
#include <stdint.h>

// ---------------------------------------------------------------------------
// PointNet++ backbone for gfx1250 (MI455X).  All MLP GEMMs run on
// v_wmma_f32_16x16x32_f16 (f16 inputs, f32 accumulate).
// ---------------------------------------------------------------------------

typedef __attribute__((ext_vector_type(16))) _Float16 v16h;
typedef __attribute__((ext_vector_type(8)))  float    v8f;

#define DEVF __device__ __forceinline__

// ---- WMMA fragment loaders (layouts per CDNA5 ISA 7.12.2) -----------------

// A 16x32 f16 tile from row-major f16 buffer (stride in halfs, must be even).
DEVF v16h load_a_frag(const uint32_t* a32, int rowBase, int strideH, int kc, int lane) {
  int row = rowBase + (lane & 15);
  int g   = lane >> 4;
  union { v16h h; uint32_t u[8]; } r;
#pragma unroll
  for (int v = 0; v < 8; ++v) {
    int kd = ((v < 4) ? 0 : 16) + g * 8 + (v & 3) * 2;      // even K offset
    r.u[v] = a32[(row * strideH + kc * 32 + kd) >> 1];
  }
  return r.h;
}

// B 32x16 f16 tile from packed weights: dword d = (k/2)*COUT + n holds
// {W[k][n], W[k+1][n]} in {lo,hi} halves.
DEVF v16h load_b_frag(const uint32_t* wp, int COUT, int kc, int nt, int lane) {
  int n  = nt * 16 + (lane & 15);
  int kb = kc * 16 + (lane >> 4) * 4;                        // dword K index
  union { v16h h; uint32_t u[8]; } r;
#pragma unroll
  for (int v = 0; v < 8; ++v) r.u[v] = wp[(size_t)(kb + ((v < 4) ? 0 : 8) + (v & 3)) * COUT + n];
  return r.h;
}

// One pointwise MLP layer on NSROWS=MT*16 rows: out = relu(in @ W + b), f16 out.
template <int MT>
DEVF void mlp_layer(const _Float16* actIn, int KP, _Float16* actOut, int COUT,
                    const uint32_t* wp, const float* bias, int lane) {
  const uint32_t* a32 = (const uint32_t*)actIn;
  const v8f zero = {0.f, 0.f, 0.f, 0.f, 0.f, 0.f, 0.f, 0.f};
  for (int nt = 0; nt < COUT / 16; ++nt) {
    v8f acc[MT];
#pragma unroll
    for (int m = 0; m < MT; ++m) acc[m] = zero;
    for (int kc = 0; kc < KP / 32; ++kc) {
      v16h bf = load_b_frag(wp, COUT, kc, nt, lane);
#pragma unroll
      for (int m = 0; m < MT; ++m) {
        v16h af = load_a_frag(a32, m * 16, KP, kc, lane);
        acc[m] = __builtin_amdgcn_wmma_f32_16x16x32_f16(
            false, af, false, bf, (short)0, acc[m], false, false);
      }
    }
    int   col = nt * 16 + (lane & 15);
    int   g   = lane >> 4;
    float bv  = bias[col];
#pragma unroll
    for (int m = 0; m < MT; ++m)
#pragma unroll
      for (int j = 0; j < 8; ++j) {
        float v = acc[m][j] + bv;
        v = v > 0.f ? v : 0.f;
        actOut[(m * 16 + j + 8 * g) * COUT + col] = (_Float16)v;
      }
  }
}

// Final MLP layer fused with max-pool over all NSROWS rows.
template <int MT>
DEVF void mlp_layer_maxpool(const _Float16* actIn, int KP, int COUT,
                            const uint32_t* wp, const float* bias, int lane,
                            float* outF, _Float16* outH) {
  const uint32_t* a32 = (const uint32_t*)actIn;
  const v8f zero = {0.f, 0.f, 0.f, 0.f, 0.f, 0.f, 0.f, 0.f};
  for (int nt = 0; nt < COUT / 16; ++nt) {
    v8f acc[MT];
#pragma unroll
    for (int m = 0; m < MT; ++m) acc[m] = zero;
    for (int kc = 0; kc < KP / 32; ++kc) {
      v16h bf = load_b_frag(wp, COUT, kc, nt, lane);
#pragma unroll
      for (int m = 0; m < MT; ++m) {
        v16h af = load_a_frag(a32, m * 16, KP, kc, lane);
        acc[m] = __builtin_amdgcn_wmma_f32_16x16x32_f16(
            false, af, false, bf, (short)0, acc[m], false, false);
      }
    }
    float bv = bias[nt * 16 + (lane & 15)];
    float mx = 0.f;                       // relu output >= 0, so 0 is identity
#pragma unroll
    for (int m = 0; m < MT; ++m)
#pragma unroll
      for (int j = 0; j < 8; ++j) {
        float v = acc[m][j] + bv;
        v = v > 0.f ? v : 0.f;
        mx = fmaxf(mx, v);
      }
    mx = fmaxf(mx, __shfl_xor(mx, 16));   // combine row halves (lanes ^16)
    if ((lane >> 4) == 0) {
      int col = nt * 16 + lane;
      outF[col] = mx;
      outH[col] = (_Float16)mx;
    }
  }
}

// ---------------------------------------------------------------------------
// Geometry kernels
// ---------------------------------------------------------------------------

__global__ void split_kernel(const float* __restrict__ pc, float* __restrict__ xyz,
                             _Float16* __restrict__ ft, int total) {
  int e = blockIdx.x * blockDim.x + threadIdx.x;
  if (e >= total) return;
  const float* s = pc + (size_t)e * 6;
  xyz[e * 3 + 0] = s[0]; xyz[e * 3 + 1] = s[1]; xyz[e * 3 + 2] = s[2];
  ft[e * 3 + 0] = (_Float16)s[3]; ft[e * 3 + 1] = (_Float16)s[4]; ft[e * 3 + 2] = (_Float16)s[5];
}

__global__ void pack_weights_kernel(const float* __restrict__ w, int K, int KP,
                                    int COUT, uint32_t* __restrict__ out) {
  int e = blockIdx.x * blockDim.x + threadIdx.x;
  if (e >= (KP / 2) * COUT) return;
  int n = e % COUT, k0 = (e / COUT) * 2;
  union { _Float16 h[2]; uint32_t u; } r;
  r.h[0] = (k0     < K) ? (_Float16)w[(size_t)k0 * COUT + n]       : (_Float16)0.f;
  r.h[1] = (k0 + 1 < K) ? (_Float16)w[(size_t)(k0 + 1) * COUT + n] : (_Float16)0.f;
  out[e] = r.u;
}

// Farthest point sampling, one block per batch; matches reference exactly
// (dist init 1e10, first idx 0, argmax with first-index tie break).
__global__ void __launch_bounds__(1024)
fps_kernel(const float* __restrict__ xyz, int NALL, int NP, int* __restrict__ out) {
  const int T = 1024, PTS = 20;                 // covers NALL <= 20480
  int b = blockIdx.x, tid = threadIdx.x;
  const float* base = xyz + (size_t)b * NALL * 3;
  float dd[PTS];
#pragma unroll
  for (int t = 0; t < PTS; ++t) dd[t] = (tid + t * T < NALL) ? 1e10f : -1.f;

  __shared__ float s_bd[32]; __shared__ int s_bi[32];
  __shared__ float s_cx, s_cy, s_cz;
  if (tid == 0) { out[b * NP] = 0; s_cx = base[0]; s_cy = base[1]; s_cz = base[2]; }
  __syncthreads();

  for (int it = 1; it < NP; ++it) {
    float cx = s_cx, cy = s_cy, cz = s_cz;
    float bd = -1e30f; int bi = 0x7fffffff;
#pragma unroll
    for (int t = 0; t < PTS; ++t) {
      int i = tid + t * T;
      if (dd[t] >= 0.f) {
        float dx = base[i * 3 + 0] - cx, dy = base[i * 3 + 1] - cy, dz = base[i * 3 + 2] - cz;
        float d = fminf(dd[t], dx * dx + dy * dy + dz * dz);
        dd[t] = d;
        if (d > bd || (d == bd && i < bi)) { bd = d; bi = i; }
      }
    }
    for (int o = 16; o > 0; o >>= 1) {          // wave32 lex-argmax reduce
      float od = __shfl_down(bd, o); int oi = __shfl_down(bi, o);
      if (od > bd || (od == bd && oi < bi)) { bd = od; bi = oi; }
    }
    int wv = tid >> 5;
    if ((tid & 31) == 0) { s_bd[wv] = bd; s_bi[wv] = bi; }
    __syncthreads();
    if (wv == 0) {
      bd = s_bd[tid & 31]; bi = s_bi[tid & 31];
      for (int o = 16; o > 0; o >>= 1) {
        float od = __shfl_down(bd, o); int oi = __shfl_down(bi, o);
        if (od > bd || (od == bd && oi < bi)) { bd = od; bi = oi; }
      }
      if (tid == 0) {
        out[b * NP + it] = bi;
        s_cx = base[bi * 3]; s_cy = base[bi * 3 + 1]; s_cz = base[bi * 3 + 2];
      }
    }
    __syncthreads();
  }
}

__global__ void gather_xyz_kernel(const float* __restrict__ xyz, int NALL,
                                  const int* __restrict__ idx, int NP,
                                  float* __restrict__ out, int total) {
  int e = blockIdx.x * blockDim.x + threadIdx.x;
  if (e >= total) return;
  int b = e / NP;
  const float* s = xyz + ((size_t)b * NALL + idx[e]) * 3;
  out[e * 3 + 0] = s[0]; out[e * 3 + 1] = s[1]; out[e * 3 + 2] = s[2];
}

// Wave-per-center ball query: ascending-index scan + ballot compaction,
// early exit at NS hits, pad with first (lowest-index) hit.
__global__ void __launch_bounds__(128)
ball_query_kernel(const float* __restrict__ xyz, int NALL,
                  const float* __restrict__ nxyz, int NP, float r2, int NS,
                  int* __restrict__ out, int total) {
  __shared__ int sidx[4][64];
  int w = threadIdx.x >> 5, lane = threadIdx.x & 31;
  int c = blockIdx.x * 4 + w;
  if (c >= total) return;
  int b = c / NP;
  float cx = nxyz[c * 3], cy = nxyz[c * 3 + 1], cz = nxyz[c * 3 + 2];
  const float* base = xyz + (size_t)b * NALL * 3;
  int found = 0;
  for (int st = 0; st < NALL; st += 32) {
    int i = st + lane;
    bool ok = false;
    if (i < NALL) {
      float dx = base[i * 3] - cx, dy = base[i * 3 + 1] - cy, dz = base[i * 3 + 2] - cz;
      ok = (dx * dx + dy * dy + dz * dz) < r2;
    }
    unsigned mask = (unsigned)__ballot(ok);
    if (ok) {
      int pos = found + __popc(mask & ((1u << lane) - 1u));
      if (pos < NS) sidx[w][pos] = i;
    }
    found += __popc(mask);
    if (found >= NS) break;
  }
  int cnt   = found < NS ? found : NS;          // center itself always hits
  int first = sidx[w][0];
  for (int s = lane; s < NS; s += 32) out[(size_t)c * NS + s] = (s < cnt) ? sidx[w][s] : first;
}

// ---------------------------------------------------------------------------
// Fused SA stage: gather group -> 3 WMMA MLP layers -> max-pool. 1 wave/center.
// ---------------------------------------------------------------------------
template <int NS, int K0P, int CF, int C1, int C2, int C3>
__global__ void __launch_bounds__(64)
sa_fused_kernel(const float* __restrict__ xyz, int NALL,
                const _Float16* __restrict__ feats,
                const float* __restrict__ nxyz, int NP,
                const int* __restrict__ ball,
                const uint32_t* __restrict__ w1, const uint32_t* __restrict__ w2,
                const uint32_t* __restrict__ w3,
                const float* __restrict__ b1, const float* __restrict__ b2,
                const float* __restrict__ b3,
                float inv_radius,
                _Float16* __restrict__ outh, float* __restrict__ outf, int total) {
  constexpr int MT = NS / 16;
  constexpr int A0 = NS * (K0P > C2 ? K0P : C2);
  constexpr int A1 = NS * (C1 > C3 ? C1 : C3);
  __shared__ _Float16 s_a0[2][A0];
  __shared__ _Float16 s_a1[2][A1];

  int w = threadIdx.x >> 5, lane = threadIdx.x & 31;
  int c = blockIdx.x * 2 + w;
  if (c >= total) return;
  int b = c / NP;
  const int* bi = ball + (size_t)c * NS;
  float cx = nxyz[c * 3], cy = nxyz[c * 3 + 1], cz = nxyz[c * 3 + 2];

  _Float16* a0 = s_a0[w];
  _Float16* a1 = s_a1[w];
  for (int s = 0; s < NS; ++s) {
    int gi = bi[s];
    const float*    px = xyz   + ((size_t)b * NALL + gi) * 3;
    const _Float16* pf = feats + ((size_t)b * NALL + gi) * CF;
    for (int k = lane; k < K0P; k += 32) {
      _Float16 hv;
      if (k < 3) {
        float v = ((k == 0 ? px[0] - cx : (k == 1 ? px[1] - cy : px[2] - cz))) * inv_radius;
        hv = (_Float16)v;
      } else if (k < 3 + CF) {
        hv = pf[k - 3];
      } else {
        hv = (_Float16)0.f;
      }
      a0[s * K0P + k] = hv;
    }
  }

  mlp_layer<MT>(a0, K0P, a1, C1, w1, b1, lane);
  mlp_layer<MT>(a1, C1, a0, C2, w2, b2, lane);
  mlp_layer_maxpool<MT>(a0, C2, C3, w3, b3, lane,
                        outf + (size_t)c * C3, outh + (size_t)c * C3);
}

// ---------------------------------------------------------------------------
// FP stage kernels
// ---------------------------------------------------------------------------

__global__ void knn3_kernel(const float* __restrict__ xu, const float* __restrict__ xk,
                            int NU, int NK, int* __restrict__ idx3,
                            float* __restrict__ w3, int total) {
  int c = blockIdx.x * blockDim.x + threadIdx.x;
  if (c >= total) return;
  int b = c / NU;
  float ux = xu[c * 3], uy = xu[c * 3 + 1], uz = xu[c * 3 + 2];
  const float* base = xk + (size_t)b * NK * 3;
  float d0 = 1e30f, d1 = 1e30f, d2 = 1e30f; int i0 = 0, i1 = 0, i2 = 0;
  for (int j = 0; j < NK; ++j) {
    float dx = base[j * 3] - ux, dy = base[j * 3 + 1] - uy, dz = base[j * 3 + 2] - uz;
    float d = dx * dx + dy * dy + dz * dz;
    if (d < d0)      { d2 = d1; i2 = i1; d1 = d0; i1 = i0; d0 = d; i0 = j; }
    else if (d < d1) { d2 = d1; i2 = i1; d1 = d;  i1 = j; }
    else if (d < d2) { d2 = d;  i2 = j; }
  }
  float r0 = 1.f / (sqrtf(fmaxf(d0, 0.f)) + 1e-8f);
  float r1 = 1.f / (sqrtf(fmaxf(d1, 0.f)) + 1e-8f);
  float r2 = 1.f / (sqrtf(fmaxf(d2, 0.f)) + 1e-8f);
  float s = r0 + r1 + r2;
  idx3[c * 3 + 0] = i0; idx3[c * 3 + 1] = i1; idx3[c * 3 + 2] = i2;
  w3[c * 3 + 0] = r0 / s; w3[c * 3 + 1] = r1 / s; w3[c * 3 + 2] = r2 / s;
}

// H[c, :] = concat(interp(feat_k f32), feat_u f16) as f16; KT = CK + CU.
__global__ void build_h_kernel(const int* __restrict__ idx3, const float* __restrict__ w3,
                               const float* __restrict__ fk, const _Float16* __restrict__ fu,
                               int NU, int NK, int CK, int CU,
                               _Float16* __restrict__ H, int total) {
  int e = blockIdx.x * blockDim.x + threadIdx.x;
  if (e >= total) return;
  int KT = CK + CU;
  int c = e / KT, k = e % KT, b = c / NU;
  _Float16 hv;
  if (k < CK) {
    const float* fb = fk + (size_t)b * NK * CK;
    float v = w3[c * 3 + 0] * fb[(size_t)idx3[c * 3 + 0] * CK + k]
            + w3[c * 3 + 1] * fb[(size_t)idx3[c * 3 + 1] * CK + k]
            + w3[c * 3 + 2] * fb[(size_t)idx3[c * 3 + 2] * CK + k];
    hv = (_Float16)v;
  } else {
    hv = fu[(size_t)c * CU + (k - CK)];
  }
  H[(size_t)c * KT + k] = hv;
}

// Plain WMMA GEMM + bias + relu; one wave = one 16x16 output tile.
template <int KP, int COUT>
__global__ void __launch_bounds__(32)
gemm_relu_kernel(const _Float16* __restrict__ A, const uint32_t* __restrict__ wp,
                 const float* __restrict__ bias, _Float16* __restrict__ outh,
                 float* __restrict__ outf, int M) {
  int mt = blockIdx.x, nt = blockIdx.y, lane = threadIdx.x;
  const uint32_t* a32 = (const uint32_t*)(A + (size_t)mt * 16 * KP);
  v8f acc = {0.f, 0.f, 0.f, 0.f, 0.f, 0.f, 0.f, 0.f};
  for (int kc = 0; kc < KP / 32; ++kc) {
    v16h bf = load_b_frag(wp, COUT, kc, nt, lane);
    v16h af = load_a_frag(a32, 0, KP, kc, lane);
    acc = __builtin_amdgcn_wmma_f32_16x16x32_f16(false, af, false, bf, (short)0,
                                                 acc, false, false);
  }
  int   col = nt * 16 + (lane & 15);
  int   g   = lane >> 4;
  float bv  = bias[col];
#pragma unroll
  for (int j = 0; j < 8; ++j) {
    float v = acc[j] + bv;
    v = v > 0.f ? v : 0.f;
    int row = mt * 16 + j + 8 * g;
    if (row < M) {
      if (outh) outh[(size_t)row * COUT + col] = (_Float16)v;
      if (outf) outf[(size_t)row * COUT + col] = v;
    }
  }
}

// ---------------------------------------------------------------------------
// Host orchestration
// ---------------------------------------------------------------------------
extern "C" void kernel_launch(void* const* d_in, const int* in_sizes, int n_in,
                              void* d_out, int out_size, void* d_ws, size_t ws_size,
                              hipStream_t stream) {
  (void)in_sizes; (void)n_in; (void)out_size; (void)ws_size;
  constexpr int B = 4, N = 20000;

  // Input order: point_clouds; per SA layer w1,w2,w3,b1,b2,b3; per FP w1,w2,b1,b2.
  const float* pc = (const float*)d_in[0];
  const float *saw[4][3], *sab[4][3], *fpw[2][2], *fpb[2][2];
  {
    int t = 1;
    for (int l = 0; l < 4; ++l) {
      for (int j = 0; j < 3; ++j) saw[l][j] = (const float*)d_in[t++];
      for (int j = 0; j < 3; ++j) sab[l][j] = (const float*)d_in[t++];
    }
    for (int l = 0; l < 2; ++l) {
      for (int j = 0; j < 2; ++j) fpw[l][j] = (const float*)d_in[t++];
      for (int j = 0; j < 2; ++j) fpb[l][j] = (const float*)d_in[t++];
    }
  }

  char* wsb = (char*)d_ws;
  size_t off = 0;
  auto alloc = [&](size_t bytes) -> void* {
    off = (off + 255) & ~(size_t)255;
    void* p = wsb + off; off += bytes; return p;
  };

  // ---- pack all weights to f16 dword-pair layout (K padded to mult of 32)
  static const int saK[4][4]  = {{6,64,64,128},{131,128,128,256},{259,128,128,256},{259,128,128,256}};
  static const int saKP0[4]   = {32,160,288,288};
  uint32_t* sawp[4][3]; uint32_t* fpwp[2][2];
  for (int l = 0; l < 4; ++l)
    for (int j = 0; j < 3; ++j) {
      int K = saK[l][j], KP = (j == 0) ? saKP0[l] : K, C = saK[l][j + 1];
      int nd = (KP / 2) * C;
      sawp[l][j] = (uint32_t*)alloc((size_t)nd * 4);
      pack_weights_kernel<<<(nd + 255) / 256, 256, 0, stream>>>(saw[l][j], K, KP, C, sawp[l][j]);
    }
  static const int fpK[2] = {512, 256};
  for (int l = 0; l < 2; ++l)
    for (int j = 0; j < 2; ++j) {
      int K = fpK[j], C = 256, nd = (K / 2) * C;
      fpwp[l][j] = (uint32_t*)alloc((size_t)nd * 4);
      pack_weights_kernel<<<(nd + 255) / 256, 256, 0, stream>>>(fpw[l][j], K, K, C, fpwp[l][j]);
    }

  // ---- buffers
  float*    xyz0  = (float*)alloc((size_t)B * N * 3 * 4);
  _Float16* feat0 = (_Float16*)alloc((size_t)B * N * 3 * 2);
  split_kernel<<<(B * N + 255) / 256, 256, 0, stream>>>(pc, xyz0, feat0, B * N);

  static const int   NPs[4] = {2048, 1024, 512, 256};
  static const int   NSs[4] = {64, 32, 16, 16};
  static const float rad[4] = {0.2f, 0.4f, 0.8f, 1.2f};
  static const int   CO[4]  = {128, 256, 256, 256};
  int cnt[5] = {N, 2048, 1024, 512, 256};

  float* nx[5]; nx[0] = xyz0;
  for (int s = 1; s < 5; ++s) nx[s] = (float*)alloc((size_t)B * cnt[s] * 3 * 4);
  int* ball[4];
  for (int s = 0; s < 4; ++s) ball[s] = (int*)alloc((size_t)B * NPs[s] * NSs[s] * 4);
  int* fpsidx = (int*)alloc((size_t)B * 2048 * 4);
  _Float16* fh[4]; float* ff[4];
  for (int s = 0; s < 4; ++s) {
    fh[s] = (_Float16*)alloc((size_t)B * NPs[s] * CO[s] * 2);
    ff[s] = (float*)alloc((size_t)B * NPs[s] * CO[s] * 4);
  }

  // ---- geometry chain (xyz hierarchy is feature-independent)
  for (int s = 0; s < 4; ++s) {
    int NALL = cnt[s], NP = NPs[s], tot = B * NP;
    fps_kernel<<<B, 1024, 0, stream>>>(nx[s], NALL, NP, fpsidx);
    gather_xyz_kernel<<<(tot + 255) / 256, 256, 0, stream>>>(nx[s], NALL, fpsidx, NP, nx[s + 1], tot);
    ball_query_kernel<<<tot / 4, 128, 0, stream>>>(nx[s], NALL, nx[s + 1], NP,
                                                   rad[s] * rad[s], NSs[s], ball[s], tot);
  }

  // ---- fused SA MLP stages (WMMA)
  sa_fused_kernel<64, 32, 3, 64, 64, 128><<<B * 2048 / 2, 64, 0, stream>>>(
      nx[0], cnt[0], feat0, nx[1], 2048, ball[0],
      sawp[0][0], sawp[0][1], sawp[0][2], sab[0][0], sab[0][1], sab[0][2],
      1.f / rad[0], fh[0], ff[0], B * 2048);
  sa_fused_kernel<32, 160, 128, 128, 128, 256><<<B * 1024 / 2, 64, 0, stream>>>(
      nx[1], cnt[1], fh[0], nx[2], 1024, ball[1],
      sawp[1][0], sawp[1][1], sawp[1][2], sab[1][0], sab[1][1], sab[1][2],
      1.f / rad[1], fh[1], ff[1], B * 1024);
  sa_fused_kernel<16, 288, 256, 128, 128, 256><<<B * 512 / 2, 64, 0, stream>>>(
      nx[2], cnt[2], fh[1], nx[3], 512, ball[2],
      sawp[2][0], sawp[2][1], sawp[2][2], sab[2][0], sab[2][1], sab[2][2],
      1.f / rad[2], fh[2], ff[2], B * 512);
  sa_fused_kernel<16, 288, 256, 128, 128, 256><<<B * 256 / 2, 64, 0, stream>>>(
      nx[3], cnt[3], fh[2], nx[4], 256, ball[3],
      sawp[3][0], sawp[3][1], sawp[3][2], sab[3][0], sab[3][1], sab[3][2],
      1.f / rad[3], fh[3], ff[3], B * 256);

  // ---- FP stages
  int*      ki = (int*)alloc((size_t)B * 1024 * 3 * 4);
  float*    kw = (float*)alloc((size_t)B * 1024 * 3 * 4);
  _Float16* H1 = (_Float16*)alloc((size_t)B * 512 * 512 * 2);
  _Float16* M1 = (_Float16*)alloc((size_t)B * 512 * 256 * 2);
  float*    F1 = (float*)alloc((size_t)B * 512 * 256 * 4);
  _Float16* H2 = (_Float16*)alloc((size_t)B * 1024 * 512 * 2);
  _Float16* M2 = (_Float16*)alloc((size_t)B * 1024 * 256 * 2);

  // FP1: u = x3 (512), k = x4 (256), feat_k = f4 (f32), feat_u = f3 (f16)
  knn3_kernel<<<(B * 512 + 255) / 256, 256, 0, stream>>>(nx[3], nx[4], 512, 256, ki, kw, B * 512);
  build_h_kernel<<<(B * 512 * 512 + 255) / 256, 256, 0, stream>>>(
      ki, kw, ff[3], fh[2], 512, 256, 256, 256, H1, B * 512 * 512);
  gemm_relu_kernel<512, 256><<<dim3(B * 512 / 16, 16), 32, 0, stream>>>(
      H1, fpwp[0][0], fpb[0][0], M1, nullptr, B * 512);
  gemm_relu_kernel<256, 256><<<dim3(B * 512 / 16, 16), 32, 0, stream>>>(
      M1, fpwp[0][1], fpb[0][1], nullptr, F1, B * 512);

  // FP2: u = x2 (1024), k = x3 (512), feat_k = FP1 out (f32), feat_u = f2 (f16)
  knn3_kernel<<<(B * 1024 + 255) / 256, 256, 0, stream>>>(nx[2], nx[3], 1024, 512, ki, kw, B * 1024);
  build_h_kernel<<<(B * 1024 * 512 + 255) / 256, 256, 0, stream>>>(
      ki, kw, F1, fh[1], 1024, 512, 256, 256, H2, B * 1024 * 512);
  gemm_relu_kernel<512, 256><<<dim3(B * 1024 / 16, 16), 32, 0, stream>>>(
      H2, fpwp[1][0], fpb[1][0], M2, nullptr, B * 1024);
  gemm_relu_kernel<256, 256><<<dim3(B * 1024 / 16, 16), 32, 0, stream>>>(
      M2, fpwp[1][1], fpb[1][1], nullptr, (float*)d_out, B * 1024);
}
